// Cluster_assigner_73461120630868
// MI455X (gfx1250) — matrix operand order repaired and checked by reading, not compiled, not run on hardware
//
#include <hip/hip_runtime.h>
#include <hip/hip_bf16.h>
#include <math.h>

// ---------------------------------------------------------------------------
// Cluster assigner pipeline for MI455X (gfx1250, wave32, WMMA).
// GEMMs run on v_wmma_f32_16x16x32_f16 with split-f16 (hi+lo) operands for
// ~f32 accuracy at matrix-core rates. Transposed operands staged via LDS.
// ---------------------------------------------------------------------------

#define BS   32
#define SEQ  1024
#define NV   1024
#define NC   64
#define DM   512
#define LDP  40            // padded LDS row stride in halves (80B) to spread banks

typedef _Float16 h16;
typedef h16   v16h __attribute__((ext_vector_type(16)));
typedef h16   v8h  __attribute__((ext_vector_type(8)));
typedef float v8f  __attribute__((ext_vector_type(8)));
typedef float v4f  __attribute__((ext_vector_type(4)));

__device__ __forceinline__ v8f zero_v8f() {
  v8f z;
  #pragma unroll
  for (int j = 0; j < 8; j++) z[j] = 0.0f;
  return z;
}

__device__ __forceinline__ v8f wmma_f16(v16h a, v16h b, v8f c) {
  // D = A(16x32) * B(32x16) + C, f32 accumulate
  return __builtin_amdgcn_wmma_f32_16x16x32_f16(false, a, false, b, (short)0, c,
                                                false, false);
}

__device__ __forceinline__ v16h cat8(v8h a, v8h b) {
  return __builtin_shufflevector(a, b, 0,1,2,3,4,5,6,7,8,9,10,11,12,13,14,15);
}

// A fragment (16x32, M x K): lane<16 holds K{0..7,16..23}, lane>=16 K{8..15,24..31}
__device__ __forceinline__ v16h load_a_frag(const h16* row, int lane) {
  int ko = (lane & 16) ? 8 : 0;
  v8h p0 = *(const v8h*)(row + ko);
  v8h p1 = *(const v8h*)(row + ko + 16);
  return cat8(p0, p1);
}

// B fragment (32x16, K x N): lane<16 holds K{0..7,16..23} of col n, lane>=16 K{8..15,24..31}
__device__ __forceinline__ v16h load_b_frag(const h16* col, int lane) {
  int ko = (lane & 16) ? 8 : 0;
  v8h p0 = *(const v8h*)(col + ko);
  v8h p1 = *(const v8h*)(col + ko + 16);
  return cat8(p0, p1);
}

// ---------------------------------------------------------------------------
// K0: zero the atomic accumulation buffers (probsum in ws, cluster_out in out)
__global__ __launch_bounds__(256) void k_zero(float* probsum, float* cluster_out) {
  int i = blockIdx.x * 256 + threadIdx.x;
  if (i < NV * NC) probsum[i] = 0.0f;
  if (i < NC * DM) cluster_out[i] = 0.0f;
}

// ---------------------------------------------------------------------------
// K1: x_emb[b,v,d] = sum_s x[b,s,v] * W[d,s] + bias[d]   (split-f16 WMMA)
// Block: 64v x 64d tile, batch b. 4 waves, each 16v x 64d. x-tile transposed via LDS.
__global__ __launch_bounds__(128) void k_xemb(
    const float* __restrict__ x, const float* __restrict__ W,
    const float* __restrict__ bias, float* __restrict__ xemb,
    h16* __restrict__ xh, h16* __restrict__ xl) {
  __shared__ h16 Ah[64 * LDP], Al[64 * LDP], Bh[64 * LDP], Bl[64 * LDP];
  const int v0 = blockIdx.x * 64, d0 = blockIdx.y * 64, b = blockIdx.z;
  const int t = threadIdx.x, w = t >> 5, lane = t & 31, m = lane & 15;
  v8f acc[4];
  #pragma unroll
  for (int n = 0; n < 4; n++) acc[n] = zero_v8f();

  for (int s0 = 0; s0 < SEQ; s0 += 32) {
    #pragma unroll
    for (int i = 0; i < 16; i++) {              // x tile: 32s x 64v -> A^T in LDS
      int e = i * 128 + t; int ss = e >> 6, vv = e & 63;
      float xv = x[((size_t)b * SEQ + s0 + ss) * NV + v0 + vv];
      h16 hi = (h16)xv;
      Ah[vv * LDP + ss] = hi;
      Al[vv * LDP + ss] = (h16)(xv - (float)hi);
    }
    #pragma unroll
    for (int i = 0; i < 16; i++) {              // W tile: 64d x 32s
      int e = i * 128 + t; int dd = e >> 5, ss = e & 31;
      float wv = W[(size_t)(d0 + dd) * SEQ + s0 + ss];
      h16 hi = (h16)wv;
      Bh[dd * LDP + ss] = hi;
      Bl[dd * LDP + ss] = (h16)(wv - (float)hi);
    }
    __syncthreads();
    v16h ah = load_a_frag(&Ah[(w * 16 + m) * LDP], lane);
    v16h al = load_a_frag(&Al[(w * 16 + m) * LDP], lane);
    #pragma unroll
    for (int n = 0; n < 4; n++) {
      v16h bh = load_b_frag(&Bh[(n * 16 + m) * LDP], lane);
      v16h bl = load_b_frag(&Bl[(n * 16 + m) * LDP], lane);
      acc[n] = wmma_f16(ah, bh, acc[n]);
      acc[n] = wmma_f16(ah, bl, acc[n]);
      acc[n] = wmma_f16(al, bh, acc[n]);
    }
    __syncthreads();
  }
  const int mhi = (lane & 16) ? 8 : 0;
  #pragma unroll
  for (int n = 0; n < 4; n++) {
    int d = d0 + n * 16 + m;
    float bv = bias[d];
    #pragma unroll
    for (int i = 0; i < 8; i++) {
      int v = v0 + w * 16 + i + mhi;
      float val = acc[n][i] + bv;
      size_t idx = ((size_t)b * NV + v) * DM + d;
      xemb[idx] = val;
      h16 hi = (h16)val;
      xh[idx] = hi;
      xl[idx] = (h16)(val - (float)hi);
    }
  }
}

// ---------------------------------------------------------------------------
// K2: rnorm[b*NV+v] = 1/max(||x_emb[b,v,:]||, 1e-12)
__global__ __launch_bounds__(128) void k_rnorm(const float* __restrict__ xemb,
                                               float* __restrict__ rnorm) {
  int row = blockIdx.x * 4 + (threadIdx.x >> 5);
  int lane = threadIdx.x & 31;
  const float* p = xemb + (size_t)row * DM;
  float s = 0.0f;
  for (int j = lane; j < DM; j += 32) { float v = p[j]; s += v * v; }
  for (int off = 16; off; off >>= 1) s += __shfl_xor(s, off, 32);
  if (lane == 0) rnorm[row] = 1.0f / fmaxf(sqrtf(s), 1e-12f);
}

// Kc: cluster_emb norms + split-f16 copies
__global__ __launch_bounds__(128) void k_cprep(const float* __restrict__ ce,
                                               float* __restrict__ cnorm,
                                               h16* __restrict__ ch,
                                               h16* __restrict__ cl) {
  int l = blockIdx.x * 4 + (threadIdx.x >> 5);
  int lane = threadIdx.x & 31;
  const float* p = ce + (size_t)l * DM;
  float s = 0.0f;
  for (int j = lane; j < DM; j += 32) {
    float v = p[j]; s += v * v;
    h16 hi = (h16)v;
    ch[(size_t)l * DM + j] = hi;
    cl[(size_t)l * DM + j] = (h16)(v - (float)hi);
  }
  for (int off = 16; off; off >>= 1) s += __shfl_xor(s, off, 32);
  if (lane == 0) cnorm[l] = 1.0f / fmaxf(sqrtf(s), 1e-12f);
}

// ---------------------------------------------------------------------------
// K3: probsum[v,l] += (1/BS) * rnorm[b,v]*cnorm[l]* (x_emb[b,v,:]·ce[l,:])
__global__ __launch_bounds__(128) void k_prob(
    const h16* __restrict__ xh, const h16* __restrict__ xl,
    const h16* __restrict__ ch, const h16* __restrict__ cl,
    const float* __restrict__ rnorm, const float* __restrict__ cnorm,
    float* __restrict__ probsum) {
  const int v0 = blockIdx.x * 64, b = blockIdx.y;
  const int t = threadIdx.x, w = t >> 5, lane = t & 31, m = lane & 15;
  const int vrow = v0 + w * 16 + m;
  const h16* arh = xh + ((size_t)b * NV + vrow) * DM;
  const h16* arl = xl + ((size_t)b * NV + vrow) * DM;
  v8f acc[4];
  #pragma unroll
  for (int n = 0; n < 4; n++) acc[n] = zero_v8f();
  for (int k0 = 0; k0 < DM; k0 += 32) {
    v16h ah = load_a_frag(arh + k0, lane);
    v16h al = load_a_frag(arl + k0, lane);
    #pragma unroll
    for (int n = 0; n < 4; n++) {
      v16h bh = load_b_frag(ch + (size_t)(n * 16 + m) * DM + k0, lane);
      v16h bl = load_b_frag(cl + (size_t)(n * 16 + m) * DM + k0, lane);
      acc[n] = wmma_f16(ah, bh, acc[n]);
      acc[n] = wmma_f16(ah, bl, acc[n]);
      acc[n] = wmma_f16(al, bh, acc[n]);
    }
  }
  const int mhi = (lane & 16) ? 8 : 0;
  #pragma unroll
  for (int n = 0; n < 4; n++) {
    int l = n * 16 + m;
    float cn = cnorm[l] * (1.0f / BS);
    #pragma unroll
    for (int i = 0; i < 8; i++) {
      int v = v0 + w * 16 + i + mhi;
      float val = acc[n][i] * rnorm[b * NV + v] * cn;
      atomicAdd(&probsum[v * NC + l], val);
    }
  }
}

// ---------------------------------------------------------------------------
// K4: sinkhorn row-normalize + concrete-bernoulli mask + batch broadcasts
__global__ __launch_bounds__(64) void k_mask(
    const float* __restrict__ probsum, const float* __restrict__ noise,
    float* __restrict__ prob_avg, float* __restrict__ mask_ws,
    float* __restrict__ prob_pb, float* __restrict__ mask_pb) {
  __shared__ float red[64];
  int v = blockIdx.x, l = threadIdx.x;
  float q = expf(probsum[v * NC + l] * 20.0f);     // exp(p / 0.05)
  red[l] = q;
  __syncthreads();
  for (int off = 32; off; off >>= 1) {
    if (l < off) red[l] += red[l + off];
    __syncthreads();
  }
  float pa = q / red[0];
  prob_avg[v * NC + l] = pa;
  float nz = noise[v * NC + l];
  float lp = logf(pa + 1e-10f) - logf(1.0f - pa + 1e-10f);
  float ln = logf(nz) - logf(1.0f - nz);
  float mk = 1.0f / (1.0f + expf(-(lp + ln)));
  mask_ws[v * NC + l] = mk;
  for (int b = 0; b < BS; b++) {
    size_t o = ((size_t)b * NV + v) * NC + l;
    prob_pb[o] = pa;
    mask_pb[o] = mk;
  }
}

// ---------------------------------------------------------------------------
// K5: raw scores[b,l,v] = ce[l,:]·x_emb[b,v,:]
__global__ __launch_bounds__(128) void k_scores(
    const h16* __restrict__ ch, const h16* __restrict__ cl,
    const h16* __restrict__ xh, const h16* __restrict__ xl,
    float* __restrict__ scores) {
  const int v0 = blockIdx.x * 64, b = blockIdx.y;
  const int t = threadIdx.x, w = t >> 5, lane = t & 31, m = lane & 15;
  const int lrow = w * 16 + m;
  const h16* arh = ch + (size_t)lrow * DM;
  const h16* arl = cl + (size_t)lrow * DM;
  v8f acc[4];
  #pragma unroll
  for (int n = 0; n < 4; n++) acc[n] = zero_v8f();
  for (int k0 = 0; k0 < DM; k0 += 32) {
    v16h ah = load_a_frag(arh + k0, lane);
    v16h al = load_a_frag(arl + k0, lane);
    #pragma unroll
    for (int n = 0; n < 4; n++) {
      int v = v0 + n * 16 + m;
      v16h bh = load_b_frag(xh + ((size_t)b * NV + v) * DM + k0, lane);
      v16h bl = load_b_frag(xl + ((size_t)b * NV + v) * DM + k0, lane);
      acc[n] = wmma_f16(ah, bh, acc[n]);
      acc[n] = wmma_f16(ah, bl, acc[n]);
      acc[n] = wmma_f16(al, bh, acc[n]);
    }
  }
  const int mhi = (lane & 16) ? 8 : 0;
  #pragma unroll
  for (int n = 0; n < 4; n++) {
    int v = v0 + n * 16 + m;
    #pragma unroll
    for (int i = 0; i < 8; i++) {
      int l = w * 16 + i + mhi;
      scores[((size_t)b * NC + l) * NV + v] = acc[n][i];
    }
  }
}

// K5b: A = softmax_v( scale * raw * mask[v,l] ), in-place over scores rows
__global__ __launch_bounds__(256) void k_softmax(float* __restrict__ scores,
                                                 const float* __restrict__ mask_ws) {
  __shared__ float sm[256];
  const float scale = 0.044194173824159216f;   // 1/sqrt(512)
  int r = blockIdx.x, l = r & 63, t = threadIdx.x;
  float* row = scores + (size_t)r * NV;
  float vals[4], mx = -INFINITY;
  #pragma unroll
  for (int i = 0; i < 4; i++) {
    int v = t + i * 256;
    float s = row[v] * mask_ws[v * NC + l] * scale;
    vals[i] = s;
    mx = fmaxf(mx, s);
  }
  sm[t] = mx; __syncthreads();
  for (int off = 128; off; off >>= 1) {
    if (t < off) sm[t] = fmaxf(sm[t], sm[t + off]);
    __syncthreads();
  }
  mx = sm[0]; __syncthreads();
  float se = 0.0f;
  #pragma unroll
  for (int i = 0; i < 4; i++) { vals[i] = expf(vals[i] - mx); se += vals[i]; }
  sm[t] = se; __syncthreads();
  for (int off = 128; off; off >>= 1) {
    if (t < off) sm[t] += sm[t + off];
    __syncthreads();
  }
  float inv = 1.0f / sm[0];
  #pragma unroll
  for (int i = 0; i < 4; i++) row[t + i * 256] = vals[i] * inv;
}

// ---------------------------------------------------------------------------
// K6: cluster_out[l,d] += (1/BS) * sum_v A[b,l,v]*x_emb[b,v,d]  (x_emb^T via LDS)
__global__ __launch_bounds__(128) void k_attn(
    const float* __restrict__ A, const h16* __restrict__ xh,
    const h16* __restrict__ xl, float* __restrict__ cluster_out) {
  __shared__ h16 Bh[64 * LDP], Bl[64 * LDP];
  const int d0 = blockIdx.x * 64, b = blockIdx.y;
  const int t = threadIdx.x, w = t >> 5, lane = t & 31, m = lane & 15;
  const float* arow = A + ((size_t)b * NC + (w * 16 + m)) * NV;
  v8f acc[4];
  #pragma unroll
  for (int n = 0; n < 4; n++) acc[n] = zero_v8f();

  for (int v0 = 0; v0 < NV; v0 += 32) {
    #pragma unroll
    for (int i = 0; i < 16; i++) {              // x_emb tile 32v x 64d -> LDS^T
      int e = i * 128 + t; int vv = e >> 6, dd = e & 63;
      size_t idx = ((size_t)b * NV + v0 + vv) * DM + d0 + dd;
      Bh[dd * LDP + vv] = xh[idx];
      Bl[dd * LDP + vv] = xl[idx];
    }
    __syncthreads();
    int koA = (lane & 16) ? 8 : 0;
    v4f q0 = *(const v4f*)(arow + v0 + koA);
    v4f q1 = *(const v4f*)(arow + v0 + koA + 4);
    v4f q2 = *(const v4f*)(arow + v0 + koA + 16);
    v4f q3 = *(const v4f*)(arow + v0 + koA + 20);
    float fa[16];
    #pragma unroll
    for (int j = 0; j < 4; j++) { fa[j] = q0[j]; fa[4+j] = q1[j]; fa[8+j] = q2[j]; fa[12+j] = q3[j]; }
    v16h ah, al;
    #pragma unroll
    for (int j = 0; j < 16; j++) {
      h16 hi = (h16)fa[j];
      ah[j] = hi;
      al[j] = (h16)(fa[j] - (float)hi);
    }
    #pragma unroll
    for (int n = 0; n < 4; n++) {
      v16h bh = load_b_frag(&Bh[(n * 16 + m) * LDP], lane);
      v16h bl = load_b_frag(&Bl[(n * 16 + m) * LDP], lane);
      acc[n] = wmma_f16(ah, bh, acc[n]);
      acc[n] = wmma_f16(ah, bl, acc[n]);
      acc[n] = wmma_f16(al, bh, acc[n]);
    }
    __syncthreads();
  }
  const int mhi = (lane & 16) ? 8 : 0;
  #pragma unroll
  for (int n = 0; n < 4; n++) {
    int d = d0 + n * 16 + m;
    #pragma unroll
    for (int i = 0; i < 8; i++) {
      int l = w * 16 + i + mhi;
      atomicAdd(&cluster_out[l * DM + d], acc[n][i] * (1.0f / BS));
    }
  }
}

// ---------------------------------------------------------------------------
extern "C" void kernel_launch(void* const* d_in, const int* in_sizes, int n_in,
                              void* d_out, int out_size, void* d_ws, size_t ws_size,
                              hipStream_t stream) {
  (void)in_sizes; (void)n_in; (void)out_size; (void)ws_size;
  const float* x     = (const float*)d_in[0];
  const float* ce    = (const float*)d_in[1];
  const float* W     = (const float*)d_in[2];
  const float* bias  = (const float*)d_in[3];
  const float* noise = (const float*)d_in[4];

  float* out        = (float*)d_out;
  float* prob_avg   = out;                                   // 1024*64
  float* cluster_o  = out + (size_t)NV * NC;                 // 64*512
  float* prob_pb    = cluster_o + (size_t)NC * DM;           // 32*1024*64
  float* mask_pb    = prob_pb + (size_t)BS * NV * NC;        // 32*1024*64
  float* xemb       = mask_pb + (size_t)BS * NV * NC;        // 32*1024*512

  char* ws = (char*)d_ws;
  const size_t XE = (size_t)BS * NV * DM;
  h16*   xh      = (h16*)ws;                       ws += XE * sizeof(h16);
  h16*   xl      = (h16*)ws;                       ws += XE * sizeof(h16);
  float* scores  = (float*)ws;                     ws += (size_t)BS * NC * NV * sizeof(float);
  float* probsum = (float*)ws;                     ws += (size_t)NV * NC * sizeof(float);
  float* rnorm   = (float*)ws;                     ws += (size_t)BS * NV * sizeof(float);
  float* cnorm   = (float*)ws;                     ws += 64 * sizeof(float);
  h16*   ch      = (h16*)ws;                       ws += (size_t)NC * DM * sizeof(h16);
  h16*   cl      = (h16*)ws;                       ws += (size_t)NC * DM * sizeof(h16);
  float* mask_ws = (float*)ws;

  k_zero   <<<dim3((NV * NC + 255) / 256), 256, 0, stream>>>(probsum, cluster_o);
  k_xemb   <<<dim3(NV / 64, DM / 64, BS), 128, 0, stream>>>(x, W, bias, xemb, xh, xl);
  k_rnorm  <<<dim3(BS * NV / 4), 128, 0, stream>>>(xemb, rnorm);
  k_cprep  <<<dim3(NC / 4), 128, 0, stream>>>(ce, cnorm, ch, cl);
  k_prob   <<<dim3(NV / 64, BS), 128, 0, stream>>>(xh, xl, ch, cl, rnorm, cnorm, probsum);
  k_mask   <<<dim3(NV), 64, 0, stream>>>(probsum, noise, prob_avg, mask_ws, prob_pb, mask_pb);
  k_scores <<<dim3(NV / 64, BS), 128, 0, stream>>>(ch, cl, xh, xl, scores);
  k_softmax<<<dim3(BS * NC), 256, 0, stream>>>(scores, mask_ws);
  k_attn   <<<dim3(DM / 64, BS), 128, 0, stream>>>(scores, xh, xl, cluster_o);
}
